// QuasiLSTMLM_70489003261963
// MI455X (gfx1250) — compile-verified, matrix operand order checked by reading, not run on hardware
//
#include <hip/hip_runtime.h>
#include <hip/hip_bf16.h>
#include <stdint.h>
#include <stddef.h>

#define D_LAYERS 8
#define DIMN     1024
#define FFN      4096
#define VOCAB    32000
#define BATCH    2
#define SEQ      2048
#define NTOK     (BATCH*SEQ)      /* 4096 tokens */
#define NCHUNK   16
#define CHUNK    (SEQ/NCHUNK)     /* 128 */

#define TM  64
#define TN  256
#define TKK 32

typedef __bf16 v16bf __attribute__((ext_vector_type(16)));
typedef unsigned short v8u16 __attribute__((ext_vector_type(8)));
typedef float v8f __attribute__((ext_vector_type(8)));
typedef unsigned int u32x4 __attribute__((ext_vector_type(4)));
typedef int i32x4 __attribute__((ext_vector_type(4)));
typedef int i32x8 __attribute__((ext_vector_type(8)));

#if defined(__has_builtin)
#if __has_builtin(__builtin_amdgcn_tensor_load_to_lds) && __has_builtin(__builtin_amdgcn_s_wait_tensorcnt)
#define USE_TDM 1
#endif
#endif
#ifndef USE_TDM
#define USE_TDM 0
#endif

__device__ __forceinline__ unsigned short f2bf(float f) {
  unsigned int u = __float_as_uint(f);
  unsigned int r = u + 0x7FFFu + ((u >> 16) & 1u);
  return (unsigned short)(r >> 16);
}
// pack two f32 -> {bf16(lo) [15:0], bf16(hi) [31:16]} in ONE v_perm_b32 (truncate)
__device__ __forceinline__ unsigned int pack2bf(float lo, float hi) {
  return __builtin_amdgcn_perm(__float_as_uint(hi), __float_as_uint(lo), 0x07060302u);
}
__device__ __forceinline__ float sigm(float x) { return 1.f / (1.f + __expf(-x)); }

__device__ __forceinline__ v16bf frag_from_lds(const unsigned short* base) {
  union { v8u16 h[2]; v16bf v; } u;
  u.h[0] = *(const v8u16*)(base);        // K = kb .. kb+7
  u.h[1] = *(const v8u16*)(base + 16);   // K = kb+16 .. kb+23
  return u.v;
}

#if USE_TDM
// 2D TDM descriptor load: tile (tile1 rows x tile0 bf16 elems), row stride
// stride0 elems, packed contiguously into LDS at lds_byte.
__device__ __forceinline__ void tdm_load_tile_bf16(
    unsigned int lds_byte, const unsigned short* g,
    unsigned int tdim0, unsigned int tdim1, unsigned long long stride0,
    unsigned int tile0, unsigned int tile1)
{
  unsigned long long ga = (unsigned long long)(uintptr_t)g;
  u32x4 g0;
  g0[0] = 1u;                                              // count=1
  g0[1] = lds_byte;                                        // lds_addr
  g0[2] = (unsigned int)ga;                                // global_addr[31:0]
  g0[3] = (unsigned int)((ga >> 32) & 0x01FFFFFFu) | (2u << 30);  // addr[56:32], type=2
  i32x8 g1;
  g1[0] = (int)(1u << 16);                                 // data_size = 2 bytes
  g1[1] = (int)((tdim0 & 0xFFFFu) << 16);                  // tensor_dim0[15:0] @ bit48
  g1[2] = (int)((tdim0 >> 16) | ((tdim1 & 0xFFFFu) << 16));
  g1[3] = (int)(((tdim1 >> 16) & 0xFFFFu) | (tile0 << 16)); // tile_dim0
  g1[4] = (int)tile1;                                      // tile_dim1 (tile_dim2=0)
  g1[5] = (int)(unsigned int)(stride0 & 0xFFFFFFFFull);    // tensor_dim0_stride lo
  g1[6] = (int)(unsigned int)((stride0 >> 32) & 0xFFFFull);
  g1[7] = 0;
  i32x4 z4 = {0, 0, 0, 0};
#if __clang_major__ >= 23
  i32x8 z8 = {0, 0, 0, 0, 0, 0, 0, 0};
  __builtin_amdgcn_tensor_load_to_lds(g0, g1, z4, z4, z8, 0);
#else
  __builtin_amdgcn_tensor_load_to_lds(g0, g1, z4, z4, 0);
#endif
}
#endif

// ---------------------------------------------------------------------------
// C[M,N] = A_bf16[M,K] * W_f32[N,K]^T + bias[N] (+ addsrc)
// Block 64(M) x 256(N), K-step 32, double-buffered LDS.
// 8 waves -> 2(M) x 4(N); wave tile 32x64 = 2x4 v_wmma_f32_16x16x32_bf16.
// A tile staged by TDM (tensor_load_to_lds), W converted f32->bf16 via v_perm.
// Requires M%64==0, N%256==0, K%32==0 (true for all GEMMs here).
// ---------------------------------------------------------------------------
__global__ __launch_bounds__(256)
void gemm_bf16_f32(const unsigned short* __restrict__ A,
                   const float* __restrict__ W,
                   const float* __restrict__ bias,
                   const float* __restrict__ addsrc,
                   float* __restrict__ C,
                   int M, int N, int K)
{
  __shared__ unsigned short sA[2][TM * TKK];    // 2 x 4 KB
  __shared__ unsigned short sB[2][TN * TKK];    // 2 x 16 KB

  const int tid  = threadIdx.x;
  const int lane = tid & 31;
  const int wave = tid >> 5;
  const int m0 = blockIdx.y * TM;
  const int n0 = blockIdx.x * TN;
  const int wm = (wave & 1) * 32;
  const int wn = (wave >> 1) * 64;
  const int fr = lane & 15;
  const int kb = (lane >> 4) * 8;

  v8f acc[2][4] = {};

  const float* wrow = W + (size_t)(n0 + tid) * K;   // one W row per thread
#if !USE_TDM
  const int arow = tid >> 2, aseg = (tid & 3) * 8;
  const unsigned short* aptr = A + (size_t)(m0 + arow) * K + aseg;
#endif

  auto stage = [&](int k0, int buf) {
    // ---- A tile: 64 x 32 bf16 ----
#if USE_TDM
    if (wave == 0) {
      tdm_load_tile_bf16((unsigned int)(uintptr_t)&sA[buf][0],
                         A + (size_t)m0 * K + k0,
                         (unsigned int)(K - k0), (unsigned int)(M - m0),
                         (unsigned long long)K, TKK, TM);
    }
#else
    *(v8u16*)&sA[buf][arow * TKK + aseg] = *(const v8u16*)(aptr + k0);
#endif
    // ---- B tile: each thread converts one 32-float W row to bf16 ----
    const float* wp = wrow + k0;
    unsigned int* sb = (unsigned int*)&sB[buf][tid * TKK];
#pragma unroll
    for (int q = 0; q < 4; q++) {
      float4 a = *(const float4*)(wp + q * 8);
      float4 b = *(const float4*)(wp + q * 8 + 4);
      uint4 v;
      v.x = pack2bf(a.x, a.y);
      v.y = pack2bf(a.z, a.w);
      v.z = pack2bf(b.x, b.y);
      v.w = pack2bf(b.z, b.w);
      *(uint4*)(sb + q * 4) = v;
    }
    if (k0 + TKK < K) __builtin_prefetch(wp + TKK, 0, 3);
  };

  stage(0, 0);
#if USE_TDM
  __builtin_amdgcn_s_wait_tensorcnt(0);
#endif
  __syncthreads();

  const int nk = K / TKK;
  for (int ki = 0; ki < nk; ki++) {
    const int cur = ki & 1;
    if (ki + 1 < nk) stage((ki + 1) * TKK, cur ^ 1);

    v16bf af[2], bfr[4];
#pragma unroll
    for (int i = 0; i < 2; i++)
      af[i] = frag_from_lds(&sA[cur][(wm + i * 16 + fr) * TKK + kb]);
#pragma unroll
    for (int j = 0; j < 4; j++)
      bfr[j] = frag_from_lds(&sB[cur][(wn + j * 16 + fr) * TKK + kb]);
#pragma unroll
    for (int i = 0; i < 2; i++)
#pragma unroll
      for (int j = 0; j < 4; j++)
        acc[i][j] = __builtin_amdgcn_wmma_f32_16x16x32_bf16(
            false, af[i], false, bfr[j], (short)0, acc[i][j], false, false);

#if USE_TDM
    __builtin_amdgcn_s_wait_tensorcnt(0);
#endif
    __syncthreads();
  }

  // Epilogue per ISA C/D layout: lanes 0-15 -> M=r, lanes 16-31 -> M=r+8; N=lane&15
  const int cn = lane & 15;
  const int cm = (lane >> 4) * 8;
#pragma unroll
  for (int i = 0; i < 2; i++) {
#pragma unroll
    for (int j = 0; j < 4; j++) {
      int mb = m0 + wm + i * 16 + cm;
      int nb = n0 + wn + j * 16 + cn;
      float bv = bias ? bias[nb] : 0.f;
#pragma unroll
      for (int r = 0; r < 8; r++) {
        size_t off = (size_t)(mb + r) * N + nb;
        float v = acc[i][j][r] + bv;
        if (addsrc) v += addsrc[off];
        C[off] = v;
      }
    }
  }
}

// ---------------------------------------------------------------------------
// Embedding gather with max_norm=1 renorm
// ---------------------------------------------------------------------------
__global__ __launch_bounds__(256)
void embed_kernel(const int* __restrict__ tokens, const float* __restrict__ emb,
                  float* __restrict__ x)
{
  int row = blockIdx.x, tid = threadIdx.x;
  const float* e = emb + (size_t)tokens[row] * DIMN;
  float ss = 0.f;
  for (int i = tid; i < DIMN; i += 256) { float v = e[i]; ss += v * v; }
  for (int off = 16; off; off >>= 1) ss += __shfl_xor(ss, off, 32);
  __shared__ float red[8];
  if ((tid & 31) == 0) red[tid >> 5] = ss;
  __syncthreads();
  float tot = 0.f;
#pragma unroll
  for (int i = 0; i < 8; i++) tot += red[i];
  float nrm = sqrtf(tot);
  float sc = (nrm > 1.f) ? 1.f / (nrm + 1e-7f) : 1.f;
  for (int i = tid; i < DIMN; i += 256) x[(size_t)row * DIMN + i] = e[i] * sc;
}

// ---------------------------------------------------------------------------
// RMSNorm -> bf16 activations (GEMM A input); RNE conversion for accuracy
// ---------------------------------------------------------------------------
__global__ __launch_bounds__(256)
void rmsnorm_bf16_kernel(const float* __restrict__ x, const float* __restrict__ w,
                         unsigned short* __restrict__ out)
{
  int row = blockIdx.x, tid = threadIdx.x;
  const float* xr = x + (size_t)row * DIMN;
  float ss = 0.f;
  for (int i = tid; i < DIMN; i += 256) { float v = xr[i]; ss += v * v; }
  for (int off = 16; off; off >>= 1) ss += __shfl_xor(ss, off, 32);
  __shared__ float red[8];
  if ((tid & 31) == 0) red[tid >> 5] = ss;
  __syncthreads();
  float tot = 0.f;
#pragma unroll
  for (int i = 0; i < 8; i++) tot += red[i];
  float sc = rsqrtf(tot / (float)DIMN + 1e-6f);
  for (int i = tid; i < DIMN; i += 256)
    out[(size_t)row * DIMN + i] = f2bf(xr[i] * sc * w[i]);
}

// gate pre-activation -> f = sigmoid(gf); v = tanh(gi)*sigmoid(gig)  (in place)
__global__ void gate_fv_kernel(float* gf, float* gi, const float* __restrict__ gig, int n)
{
  int i = blockIdx.x * 256 + threadIdx.x;
  if (i < n) { gf[i] = sigm(gf[i]); gi[i] = tanhf(gi[i]) * sigm(gig[i]); }
}

// ---------------------------------------------------------------------------
// Chunked linear scan  h_t = f_t h_{t-1} + v_t,  h_{-1} = h0[d]
// ---------------------------------------------------------------------------
__global__ void scan_pass1(const float* __restrict__ F, const float* __restrict__ Vv,
                           float* __restrict__ P, float* __restrict__ Hc)
{
  int tid = blockIdx.x * 256 + threadIdx.x;         // BATCH*DIM*NCHUNK
  if (tid >= BATCH * DIMN * NCHUNK) return;
  int d = tid % DIMN, r = tid / DIMN;
  int b = r % BATCH, c = r / BATCH;
  float p = 1.f, h = 0.f;
  int base = (b * SEQ + c * CHUNK) * DIMN + d;
  for (int t = 0; t < CHUNK; t++) {
    float f = F[base + t * DIMN], v = Vv[base + t * DIMN];
    p *= f; h = f * h + v;
  }
  P[tid] = p; Hc[tid] = h;
}

__global__ void scan_pass2(const float* __restrict__ P, const float* __restrict__ Hc,
                           const float* __restrict__ h0, float* __restrict__ St)
{
  int tid = blockIdx.x * 256 + threadIdx.x;         // BATCH*DIM
  if (tid >= BATCH * DIMN) return;
  int d = tid % DIMN;
  float h = h0[d];
  for (int c = 0; c < NCHUNK; c++) {
    int idx = c * BATCH * DIMN + tid;
    St[idx] = h;
    h = P[idx] * h + Hc[idx];
  }
}

__global__ void scan_pass3(const float* __restrict__ F, const float* __restrict__ Vv,
                           const float* __restrict__ St, float* __restrict__ H)
{
  int tid = blockIdx.x * 256 + threadIdx.x;
  if (tid >= BATCH * DIMN * NCHUNK) return;
  int d = tid % DIMN, r = tid / DIMN;
  int b = r % BATCH, c = r / BATCH;
  float h = St[tid];
  int base = (b * SEQ + c * CHUNK) * DIMN + d;
  for (int t = 0; t < CHUNK; t++) {
    h = F[base + t * DIMN] * h + Vv[base + t * DIMN];
    H[base + t * DIMN] = h;
  }
}

// x += tanh(h) * sigmoid(og)
__global__ void out_gate_kernel(float* __restrict__ x, const float* __restrict__ H,
                                const float* __restrict__ OG, int n)
{
  int i = blockIdx.x * 256 + threadIdx.x;
  if (i < n) x[i] += tanhf(H[i]) * sigm(OG[i]);
}

// u = uc * silu(ua) -> bf16
__global__ void swiglu_kernel(const float* __restrict__ uc, const float* __restrict__ ua,
                              unsigned short* __restrict__ ub, int n)
{
  int i = blockIdx.x * 256 + threadIdx.x;
  if (i < n) { float a = ua[i]; ub[i] = f2bf(uc[i] * a * sigm(a)); }
}

// ---------------------------------------------------------------------------
extern "C" void kernel_launch(void* const* d_in, const int* in_sizes, int n_in,
                              void* d_out, int out_size, void* d_ws, size_t ws_size,
                              hipStream_t stream)
{
  (void)in_sizes; (void)n_in; (void)out_size; (void)ws_size;
  const int*   tokens = (const int*)  d_in[0];
  const float* emb    = (const float*)d_in[1];
  const float* h0     = (const float*)d_in[2];
  const float* Wf  = (const float*)d_in[3];  const float* bfo_ = (const float*)d_in[4];
  const float* Wi  = (const float*)d_in[5];  const float* bi   = (const float*)d_in[6];
  const float* Wig = (const float*)d_in[7];  const float* big  = (const float*)d_in[8];
  const float* Wog = (const float*)d_in[9];  const float* bog  = (const float*)d_in[10];
  const float* n1  = (const float*)d_in[11]; const float* n2   = (const float*)d_in[12];
  const float* Wfc = (const float*)d_in[13]; const float* bfc  = (const float*)d_in[14];
  const float* Wfa = (const float*)d_in[15]; const float* bfa  = (const float*)d_in[16];
  const float* Wfo = (const float*)d_in[17]; const float* bff  = (const float*)d_in[18];
  const float* nl  = (const float*)d_in[19];
  const float* Wout= (const float*)d_in[20]; const float* bout = (const float*)d_in[21];
  float* out = (float*)d_out;

  char* ws = (char*)d_ws;
  size_t off = 0;
  auto alloc = [&](size_t bytes) -> char* {
    char* p = ws + off; off += (bytes + 255) & ~(size_t)255; return p;
  };
  float*          x   = (float*)         alloc((size_t)NTOK * DIMN * 4);
  unsigned short* Abf = (unsigned short*)alloc((size_t)NTOK * FFN * 2);
  float* b0 = (float*)alloc((size_t)NTOK * DIMN * 4);
  float* b1 = (float*)alloc((size_t)NTOK * DIMN * 4);
  float* b2 = (float*)alloc((size_t)NTOK * DIMN * 4);
  float* b3 = (float*)alloc((size_t)NTOK * DIMN * 4);
  float* uc = (float*)alloc((size_t)NTOK * FFN * 4);
  float* ua = (float*)alloc((size_t)NTOK * FFN * 4);
  float* cP = (float*)alloc((size_t)BATCH * DIMN * NCHUNK * 4);
  float* cH = (float*)alloc((size_t)BATCH * DIMN * NCHUNK * 4);
  float* cS = (float*)alloc((size_t)BATCH * DIMN * NCHUNK * 4);

  dim3 blk(256);
  const int nED = NTOK * DIMN;
  const int nEF = NTOK * FFN;
  const int nSC = BATCH * DIMN * NCHUNK;
  dim3 gDIM(DIMN / TN, NTOK / TM);
  dim3 gFF (FFN  / TN, NTOK / TM);
  dim3 gV  (VOCAB/ TN, NTOK / TM);

  embed_kernel<<<NTOK, blk, 0, stream>>>(tokens, emb, x);

  for (int l = 0; l < D_LAYERS; l++) {
    const float* Wf_l  = Wf  + (size_t)l * DIMN * DIMN;
    const float* Wi_l  = Wi  + (size_t)l * DIMN * DIMN;
    const float* Wig_l = Wig + (size_t)l * DIMN * DIMN;
    const float* Wog_l = Wog + (size_t)l * DIMN * DIMN;
    const float* Wfc_l = Wfc + (size_t)l * FFN * DIMN;
    const float* Wfa_l = Wfa + (size_t)l * FFN * DIMN;
    const float* Wfo_l = Wfo + (size_t)l * DIMN * FFN;

    // --- QuasiLSTM sub-block ---
    rmsnorm_bf16_kernel<<<NTOK, blk, 0, stream>>>(x, n1 + (size_t)l * DIMN, Abf);
    gemm_bf16_f32<<<gDIM, blk, 0, stream>>>(Abf, Wf_l,  bfo_ + (size_t)l * DIMN, nullptr, b0, NTOK, DIMN, DIMN);
    gemm_bf16_f32<<<gDIM, blk, 0, stream>>>(Abf, Wi_l,  bi   + (size_t)l * DIMN, nullptr, b1, NTOK, DIMN, DIMN);
    gemm_bf16_f32<<<gDIM, blk, 0, stream>>>(Abf, Wig_l, big  + (size_t)l * DIMN, nullptr, b2, NTOK, DIMN, DIMN);
    gate_fv_kernel<<<(nED + 255) / 256, blk, 0, stream>>>(b0, b1, b2, nED);
    scan_pass1<<<(nSC + 255) / 256, blk, 0, stream>>>(b0, b1, cP, cH);
    scan_pass2<<<(BATCH * DIMN + 255) / 256, blk, 0, stream>>>(cP, cH, h0 + (size_t)l * DIMN, cS);
    scan_pass3<<<(nSC + 255) / 256, blk, 0, stream>>>(b0, b1, cS, b2);  // h -> b2
    gemm_bf16_f32<<<gDIM, blk, 0, stream>>>(Abf, Wog_l, bog + (size_t)l * DIMN, nullptr, b3, NTOK, DIMN, DIMN);
    out_gate_kernel<<<(nED + 255) / 256, blk, 0, stream>>>(x, b2, b3, nED);

    // --- SwiGLU FFN sub-block ---
    rmsnorm_bf16_kernel<<<NTOK, blk, 0, stream>>>(x, n2 + (size_t)l * DIMN, Abf);
    gemm_bf16_f32<<<gFF, blk, 0, stream>>>(Abf, Wfc_l, bfc + (size_t)l * FFN, nullptr, uc, NTOK, FFN, DIMN);
    gemm_bf16_f32<<<gFF, blk, 0, stream>>>(Abf, Wfa_l, bfa + (size_t)l * FFN, nullptr, ua, NTOK, FFN, DIMN);
    swiglu_kernel<<<(nEF + 255) / 256, blk, 0, stream>>>(uc, ua, Abf, nEF);
    gemm_bf16_f32<<<gDIM, blk, 0, stream>>>(Abf, Wfo_l, bff + (size_t)l * DIMN, x, x, NTOK, DIMN, FFN);
  }

  rmsnorm_bf16_kernel<<<NTOK, blk, 0, stream>>>(x, nl, Abf);
  gemm_bf16_f32<<<gV, blk, 0, stream>>>(Abf, Wout, bout, nullptr, out, NTOK, VOCAB, DIMN);
}
